// MultiHeadAttention_28948079575572
// MI455X (gfx1250) — matrix operand order repaired and checked by reading, not compiled, not run on hardware
//
#include <hip/hip_runtime.h>
#include <hip/hip_bf16.h>

#define N_TOK   4096
#define FEAT    256
#define OUT_D   512
#define HEADS   8
#define HEAD_DIM 64
#define BATCH_N 8

typedef __attribute__((ext_vector_type(16))) _Float16 v16h;
typedef __attribute__((ext_vector_type(8)))  _Float16 v8h;
typedef __attribute__((ext_vector_type(8)))  float    v8f;

static __device__ __forceinline__ v8f wmma16(v16h a, v16h b, v8f c) {
  // D = A(16x32 f16) * B(32x16 f16) + C(16x16 f32)
  return __builtin_amdgcn_wmma_f32_16x16x32_f16(false, a, false, b, (short)0, c,
                                                false, false);
}
static __device__ __forceinline__ v16h vcat(v8h lo, v8h hi) {
  return __builtin_shufflevector(lo, hi, 0,1,2,3,4,5,6,7,8,9,10,11,12,13,14,15);
}

// CDNA5 async global->LDS copy (ASYNCcnt-tracked, no VGPR round trip).
// LDS operand: low 32 bits of the generic pointer == LDS offset (ISA 10.2).
static __device__ __forceinline__ void async_copy_b128(const void* g, void* l) {
  unsigned lds = (unsigned)(uintptr_t)l;
  asm volatile("global_load_async_to_lds_b128 %0, %1, off"
               :: "v"(lds), "v"(g) : "memory");
}
static __device__ __forceinline__ void wait_async0() {
  asm volatile("s_wait_asynccnt 0" ::: "memory");
}

// ---------------------------------------------------------------------------
// Convert weights fp32 [K,Nout] -> f16 transposed [Nout,K] (done once; makes
// GEMM B tiles contiguous so they can be staged with async LDS copies).
// ---------------------------------------------------------------------------
__global__ __launch_bounds__(256) void wcvt_kernel(const float* __restrict__ W,
                                                   _Float16* __restrict__ Wt,
                                                   int K, int Nout) {
  int idx = blockIdx.x * 256 + threadIdx.x;
  if (idx < K * Nout) {
    int k = idx / Nout, n = idx - k * Nout;
    Wt[(size_t)n * K + k] = (_Float16)W[idx];
  }
}

// ---------------------------------------------------------------------------
// Tiled GEMM: out = A[M,K] @ Wt^T[K,Nout] + bias, via v_wmma_f32_16x16x32_f16.
// 128x128 tile per 256-thread block (8 waves, each 32x64).
// B tiles (and A tiles when already f16) staged via global_load_async_to_lds.
// OMODE 0: f16 row-major out, scaled   (Q/K projections)
// OMODE 1: f16 transposed out [col][M] (V projection -> Vt)
// OMODE 2: f32 row-major out           (final output projection)
// ---------------------------------------------------------------------------
template <bool A_HALF, int OMODE>
__global__ __launch_bounds__(256) void gemm_wmma_kernel(
    const void* __restrict__ Av, const _Float16* __restrict__ Wt,
    const float* __restrict__ bias, void* __restrict__ outv,
    int M, int K, int Nout, float scale)
{
  __shared__ __align__(16) _Float16 As[128][40];  // 80B row keeps 16B align
  __shared__ __align__(16) _Float16 Bs[128][40];  // Bs[n][k] = Wt[n][k]

  const int t    = threadIdx.x;
  const int lane = t & 31;
  const int wave = t >> 5;
  const int wm   = (wave & 3) * 32;   // 4 waves along M
  const int wn   = (wave >> 2) * 64;  // 2 waves along N
  const int row0 = blockIdx.x * 128;
  const int col0 = blockIdx.y * 128;

  v8f zero = {};
  v8f acc[2][4];
  for (int mi = 0; mi < 2; ++mi)
    for (int ni = 0; ni < 4; ++ni) acc[mi][ni] = zero;

  for (int k0 = 0; k0 < K; k0 += 32) {
    // ---- stage A (convert fp32->f16, or async copy if already f16) ----
    if constexpr (A_HALF) {
      for (int idx = t; idx < 512; idx += 256) {
        int r = idx >> 2, c = (idx & 3) * 8;
        async_copy_b128((const _Float16*)Av + (size_t)(row0 + r) * K + k0 + c,
                        &As[r][c]);
      }
    } else {
      for (int idx = t; idx < 128 * 32; idx += 256) {
        int r = idx >> 5, c = idx & 31;
        As[r][c] = (_Float16)((const float*)Av)[(size_t)(row0 + r) * K + k0 + c];
      }
      if (k0 + 32 < K) {  // prefetch next A k-slice
        int r = t >> 1, c = (t & 1) * 16;
        __builtin_prefetch(((const float*)Av) + (size_t)(row0 + r) * K +
                               k0 + 32 + c, 0, 0);
      }
    }
    // ---- stage B via async global->LDS (f16, pre-transposed weights) ----
    for (int idx = t; idx < 512; idx += 256) {
      int n = idx >> 2, c = (idx & 3) * 8;
      async_copy_b128(Wt + (size_t)(col0 + n) * K + k0 + c, &Bs[n][c]);
    }
    wait_async0();
    __syncthreads();

    const int rA = lane & 15, hA = lane >> 4;
    v16h a[2], b[4];
    #pragma unroll
    for (int mi = 0; mi < 2; ++mi) {
      // A 16x32 layout: 16 contiguous k starting at hA*16
      const _Float16* p = &As[wm + mi * 16 + rA][hA * 16];
      a[mi] = vcat(*(const v8h*)p, *(const v8h*)(p + 8));
    }
    #pragma unroll
    for (int ni = 0; ni < 4; ++ni) {
      // B 32x16 layout: lane = column, 16 contiguous k (lanes 16-31: k+16)
      const _Float16* p = &Bs[wn + ni * 16 + rA][hA * 16];
      b[ni] = vcat(*(const v8h*)p, *(const v8h*)(p + 8));
    }
    #pragma unroll
    for (int mi = 0; mi < 2; ++mi)
      #pragma unroll
      for (int ni = 0; ni < 4; ++ni)
        acc[mi][ni] = wmma16(a[mi], b[ni], acc[mi][ni]);
    __syncthreads();
  }

  const int cC = lane & 15, hC = lane >> 4;
  #pragma unroll
  for (int mi = 0; mi < 2; ++mi)
    #pragma unroll
    for (int ni = 0; ni < 4; ++ni)
      #pragma unroll
      for (int r = 0; r < 8; ++r) {
        int row = row0 + wm + mi * 16 + hC * 8 + r;
        int col = col0 + wn + ni * 16 + cC;
        float v = acc[mi][ni][r] + bias[col];
        if (OMODE == 0)
          ((_Float16*)outv)[(size_t)row * Nout + col] = (_Float16)(v * scale);
        else if (OMODE == 1)
          ((_Float16*)outv)[(size_t)col * M + row] = (_Float16)v;
        else
          ((float*)outv)[(size_t)row * Nout + col] = v;
      }
}

// ---------------------------------------------------------------------------
// Per-batch contiguous key ranges (kv_batch is sorted).
// ---------------------------------------------------------------------------
__global__ __launch_bounds__(256) void kv_range_kernel(
    const int* __restrict__ kv_batch, int* __restrict__ kvStart,
    int* __restrict__ kvEnd)
{
  int t = threadIdx.x;
  if (t < BATCH_N) { kvStart[t] = N_TOK; kvEnd[t] = 0; }
  __syncthreads();
  for (int i = t; i < N_TOK; i += 256) {
    int b = kv_batch[i];
    atomicMin(&kvStart[b], i);
    atomicMax(&kvEnd[b], i + 1);
  }
}

// ---------------------------------------------------------------------------
// Flash attention, one wave per (16-query tile, head). Key loop restricted to
// the union of the tile's batch segments; exact mask via batch-id compare.
// ---------------------------------------------------------------------------
__global__ __launch_bounds__(32) void attn_kernel(
    const _Float16* __restrict__ Q16, const _Float16* __restrict__ K16,
    const _Float16* __restrict__ Vt16, _Float16* __restrict__ H16,
    const int* __restrict__ q_batch, const int* __restrict__ kv_batch,
    const int* __restrict__ kvStart, const int* __restrict__ kvEnd)
{
  __shared__ __align__(16) _Float16 Plds[16 * 32];

  const int lane = threadIdx.x & 31;
  const int c16  = lane & 15;
  const int half = lane >> 4;
  const int q0   = blockIdx.x * 16;
  const int h    = blockIdx.y;

  // Q A-fragments (Q already pre-scaled by 1/sqrt(Dh)); Dh=64 -> 2 fragments
  v16h aQ[2];
  {
    const _Float16* qp = Q16 + (size_t)(q0 + c16) * OUT_D + h * HEAD_DIM;
    #pragma unroll
    for (int f = 0; f < 2; ++f) {
      const _Float16* p = qp + f * 32 + half * 16;
      aQ[f] = vcat(*(const v8h*)p, *(const v8h*)(p + 8));
    }
  }
  int qb[8];
  #pragma unroll
  for (int r = 0; r < 8; ++r) qb[r] = q_batch[q0 + half * 8 + r];

  // key range for this tile (batch ids sorted -> contiguous segments)
  int bLo = q_batch[q0], bHi = q_batch[q0 + 15];
  int ks = N_TOK, ke = 0;
  for (int b = bLo; b <= bHi; ++b) {
    ks = min(ks, kvStart[b]);
    ke = max(ke, kvEnd[b]);
  }
  ks &= ~31;
  ke = (ke + 31) & ~31;
  if (ks >= ke) { ks = 0; ke = 0; }

  float mrow[8], lrow[8];
  #pragma unroll
  for (int r = 0; r < 8; ++r) { mrow[r] = -3.0e38f; lrow[r] = 0.0f; }
  v8f zero = {};
  v8f o[4];
  #pragma unroll
  for (int tt = 0; tt < 4; ++tt) o[tt] = zero;

  for (int kb = ks; kb < ke; kb += 32) {
    // S = Q K^T for 2 key blocks of 16, contraction over Dh=64 (2 WMMAs each)
    v8f s0 = zero, s1 = zero;
    #pragma unroll
    for (int f = 0; f < 2; ++f) {
      const _Float16* kp0 =
          K16 + (size_t)(kb + c16) * OUT_D + h * HEAD_DIM + f * 32 + half * 16;
      const _Float16* kp1 =
          K16 + (size_t)(kb + 16 + c16) * OUT_D + h * HEAD_DIM + f * 32 + half * 16;
      v16h b0 = vcat(*(const v8h*)kp0, *(const v8h*)(kp0 + 8));
      v16h b1 = vcat(*(const v8h*)kp1, *(const v8h*)(kp1 + 8));
      s0 = wmma16(aQ[f], b0, s0);
      s1 = wmma16(aQ[f], b1, s1);
    }
    int kbat0 = kv_batch[kb + c16];
    int kbat1 = kv_batch[kb + 16 + c16];

    float alpha[8];
    #pragma unroll
    for (int r = 0; r < 8; ++r) {
      float x0 = (qb[r] == kbat0) ? s0[r] : -1e9f;
      float x1 = (qb[r] == kbat1) ? s1[r] : -1e9f;
      float mstep = fmaxf(x0, x1);
      #pragma unroll
      for (int mm = 8; mm >= 1; mm >>= 1)
        mstep = fmaxf(mstep, __shfl_xor(mstep, mm, 32));
      float mnew = fmaxf(mrow[r], mstep);
      float a  = __expf(mrow[r] - mnew);
      float p0 = __expf(x0 - mnew);
      float p1 = __expf(x1 - mnew);
      float ps = p0 + p1;
      #pragma unroll
      for (int mm = 8; mm >= 1; mm >>= 1) ps += __shfl_xor(ps, mm, 32);
      lrow[r] = lrow[r] * a + ps;
      mrow[r] = mnew;
      alpha[r] = a;
      int row = half * 8 + r;  // C-layout row -> A-layout row-major staging
      Plds[row * 32 + c16]      = (_Float16)p0;
      Plds[row * 32 + 16 + c16] = (_Float16)p1;
    }
    #pragma unroll
    for (int tt = 0; tt < 4; ++tt)
      #pragma unroll
      for (int r = 0; r < 8; ++r) o[tt][r] *= alpha[r];
    __syncthreads();

    // P as A-fragment (16x32), V^T as B-fragments (32 keys x 16 d)
    const _Float16* pp = &Plds[c16 * 32 + half * 16];
    v16h pf = vcat(*(const v8h*)pp, *(const v8h*)(pp + 8));
    #pragma unroll
    for (int tt = 0; tt < 4; ++tt) {
      const _Float16* vp =
          Vt16 + (size_t)(h * HEAD_DIM + tt * 16 + c16) * N_TOK + kb + half * 16;
      v16h bv = vcat(*(const v8h*)vp, *(const v8h*)(vp + 8));
      o[tt] = wmma16(pf, bv, o[tt]);
    }
    __syncthreads();
  }

  float inv[8];
  #pragma unroll
  for (int r = 0; r < 8; ++r) inv[r] = lrow[r] > 0.0f ? 1.0f / lrow[r] : 0.0f;
  #pragma unroll
  for (int tt = 0; tt < 4; ++tt)
    #pragma unroll
    for (int r = 0; r < 8; ++r) {
      int row = q0 + half * 8 + r;
      int col = h * HEAD_DIM + tt * 16 + c16;
      H16[(size_t)row * OUT_D + col] = (_Float16)(o[tt][r] * inv[r]);
    }
}

// ---------------------------------------------------------------------------
extern "C" void kernel_launch(void* const* d_in, const int* in_sizes, int n_in,
                              void* d_out, int out_size, void* d_ws, size_t ws_size,
                              hipStream_t stream) {
  const float* qf       = (const float*)d_in[0];
  const float* kf       = (const float*)d_in[1];
  const float* vf       = (const float*)d_in[2];
  const int*   q_batch  = (const int*)d_in[3];
  const int*   kv_batch = (const int*)d_in[4];
  const float* Wq = (const float*)d_in[5];  const float* bq = (const float*)d_in[6];
  const float* Wk = (const float*)d_in[7];  const float* bk = (const float*)d_in[8];
  const float* Wv = (const float*)d_in[9];  const float* bv = (const float*)d_in[10];
  const float* Wo = (const float*)d_in[11]; const float* bo = (const float*)d_in[12];
  float* out = (float*)d_out;

  char* ws = (char*)d_ws;
  const size_t SZ = (size_t)N_TOK * OUT_D * sizeof(_Float16);  // 4 MB
  _Float16* Q16  = (_Float16*)(ws);
  _Float16* K16  = (_Float16*)(ws + SZ);
  _Float16* Vt16 = (_Float16*)(ws + 2 * SZ);
  _Float16* H16  = (_Float16*)(ws + 3 * SZ);
  int* kvStart   = (int*)(ws + 4 * SZ);
  int* kvEnd     = kvStart + BATCH_N;
  _Float16* WtQ  = (_Float16*)(ws + 4 * SZ + 256);
  _Float16* WtK  = WtQ + (size_t)OUT_D * FEAT;
  _Float16* WtV  = WtK + (size_t)OUT_D * FEAT;
  _Float16* WtO  = WtV + (size_t)OUT_D * FEAT;

  // One-shot weight conversion to transposed f16 (enables async B staging)
  wcvt_kernel<<<(FEAT * OUT_D) / 256, 256, 0, stream>>>(Wq, WtQ, FEAT, OUT_D);
  wcvt_kernel<<<(FEAT * OUT_D) / 256, 256, 0, stream>>>(Wk, WtK, FEAT, OUT_D);
  wcvt_kernel<<<(FEAT * OUT_D) / 256, 256, 0, stream>>>(Wv, WtV, FEAT, OUT_D);
  wcvt_kernel<<<(OUT_D * OUT_D) / 256, 256, 0, stream>>>(Wo, WtO, OUT_D, OUT_D);

  dim3 gProj(N_TOK / 128, OUT_D / 128);
  // Q scaled by 1/sqrt(64) = 0.125 so attention scores come out pre-scaled
  gemm_wmma_kernel<false, 0><<<gProj, 256, 0, stream>>>(qf, WtQ, bq, Q16,
      N_TOK, FEAT, OUT_D, 0.125f);
  gemm_wmma_kernel<false, 0><<<gProj, 256, 0, stream>>>(kf, WtK, bk, K16,
      N_TOK, FEAT, OUT_D, 1.0f);
  gemm_wmma_kernel<false, 1><<<gProj, 256, 0, stream>>>(vf, WtV, bv, Vt16,
      N_TOK, FEAT, OUT_D, 1.0f);

  kv_range_kernel<<<1, 256, 0, stream>>>(kv_batch, kvStart, kvEnd);

  attn_kernel<<<dim3(N_TOK / 16, HEADS), 32, 0, stream>>>(
      Q16, K16, Vt16, H16, q_batch, kv_batch, kvStart, kvEnd);

  gemm_wmma_kernel<true, 2><<<gProj, 256, 0, stream>>>(H16, WtO, bo, out,
      N_TOK, OUT_D, OUT_D, 1.0f);
}
